// GNNActor_31413390803533
// MI455X (gfx1250) — matrix operand order repaired and checked by reading, not compiled, not run on hardware
//
#include <hip/hip_runtime.h>
#include <hip/hip_bf16.h>
#include <math.h>

typedef __attribute__((ext_vector_type(16))) __bf16 v16bf;
typedef __attribute__((ext_vector_type(8)))  float  v8f;

#define KD 128  // inner dim is always 128

// ---------------------------------------------------------------- small utils
__global__ void k_fill1(float* __restrict__ p, int n) {
    int i = blockIdx.x * blockDim.x + threadIdx.x;
    if (i < n) p[i] = 1.0f;
}

__global__ void k_degcount(const int* __restrict__ ei, float* __restrict__ deg, int ne) {
    int e = blockIdx.x * blockDim.x + threadIdx.x;
    if (e < ne) {
        int dst = ei[ne + e];
        __hip_atomic_fetch_add(&deg[dst], 1.0f, __ATOMIC_RELAXED, __HIP_MEMORY_SCOPE_AGENT);
    }
}

__global__ void k_rsqrt(float* __restrict__ p, int n) {
    int i = blockIdx.x * blockDim.x + threadIdx.x;
    if (i < n) p[i] = rsqrtf(p[i]);
}

// ------------------------------------------------- WMMA GEMM:  Y = X @ W (+b)
// X: n x 128 f32, W: 128 x OUTC f32 (row-major), Y: n x OUTC f32.
// 256 threads = 8 waves; each wave computes 16 rows x OUTC cols.
template <int OUTC, bool ADD_BIAS, bool RELU>
__global__ void __launch_bounds__(256)
k_gemm(const float* __restrict__ X, const float* __restrict__ W,
       const float* __restrict__ bias, float* __restrict__ Y, int n) {
    // W transposed in LDS as bf16: wT[c*128 + k]
    __shared__ __bf16 wT[OUTC * KD];
    const int tid = threadIdx.x;
    // coalesced global read, transposed (strided) LDS store
    for (int s = tid; s < OUTC * KD; s += 256) {
        int k = s / OUTC;          // W row (consecutive s -> consecutive c: coalesced)
        int c = s % OUTC;
        wT[c * KD + k] = (__bf16)W[s];
    }
    __syncthreads();

    const int wave = tid >> 5;
    const int lane = tid & 31;
    const int rowBase = blockIdx.x * 128 + wave * 16;
    const int r  = rowBase + (lane & 15);          // A-matrix row for this lane
    const int rc = (r < n) ? r : (n - 1);          // clamp: garbage rows never stored
    const int hi = (lane < 16) ? 0 : 1;            // lane half

    constexpr int NT = OUTC / 16;                  // 16-col tiles
    v8f acc[NT];
#pragma unroll
    for (int t = 0; t < NT; ++t) acc[t] = (v8f){0, 0, 0, 0, 0, 0, 0, 0};

    const float* xr = X + (size_t)rc * KD;

#pragma unroll
    for (int ks = 0; ks < 4; ++ks) {               // K = 4 * 32
        // --- A fragment: lanes 0-15 hold K b..b+15; lanes 16-31 K b+16..b+31
        const int ao = ks * 32 + hi * 16;
        float4 p0 = *(const float4*)(xr + ao +  0);
        float4 p1 = *(const float4*)(xr + ao +  4);
        float4 p2 = *(const float4*)(xr + ao +  8);
        float4 p3 = *(const float4*)(xr + ao + 12);
        v16bf a;
        a[0]  = (__bf16)p0.x; a[1]  = (__bf16)p0.y; a[2]  = (__bf16)p0.z; a[3]  = (__bf16)p0.w;
        a[4]  = (__bf16)p1.x; a[5]  = (__bf16)p1.y; a[6]  = (__bf16)p1.z; a[7]  = (__bf16)p1.w;
        a[8]  = (__bf16)p2.x; a[9]  = (__bf16)p2.y; a[10] = (__bf16)p2.z; a[11] = (__bf16)p2.w;
        a[12] = (__bf16)p3.x; a[13] = (__bf16)p3.y; a[14] = (__bf16)p3.z; a[15] = (__bf16)p3.w;

        // --- B fragments: lane = col, lanes 0-15 K b..b+15, lanes 16-31 K b+16..b+31
        const int kbB = ks * 32 + hi * 16;
#pragma unroll
        for (int t = 0; t < NT; ++t) {
            const int c = t * 16 + (lane & 15);
            v16bf b = *(const v16bf*)&wT[c * KD + kbB];   // 32B contiguous LDS load
            acc[t] = __builtin_amdgcn_wmma_f32_16x16x32_bf16(
                false, a, false, b, (short)0, acc[t], false, false);
        }
    }

    // --- epilogue: C/D layout — VGPR j: row j (lanes 0-15) / row j+8 (lanes 16-31)
    const int cBase = lane & 15;
    if (rowBase + 16 <= n) {
        // full tile (781 of 782 blocks): no guards at all
#pragma unroll
        for (int t = 0; t < NT; ++t) {
            const int c = t * 16 + cBase;
            float bv = ADD_BIAS ? bias[c] : 0.f;
#pragma unroll
            for (int j = 0; j < 8; ++j) {
                int rr = rowBase + j + hi * 8;
                float v = acc[t][j] + bv;
                if (RELU) v = fmaxf(v, 0.f);
                Y[(size_t)rr * OUTC + c] = v;
            }
        }
    } else {
#pragma unroll
        for (int t = 0; t < NT; ++t) {
            const int c = t * 16 + cBase;
            float bv = ADD_BIAS ? bias[c] : 0.f;
#pragma unroll
            for (int j = 0; j < 8; ++j) {
                int rr = rowBase + j + hi * 8;
                if (rr < n) {
                    float v = acc[t][j] + bv;
                    if (RELU) v = fmaxf(v, 0.f);
                    Y[(size_t)rr * OUTC + c] = v;
                }
            }
        }
    }
}

// --------------------------------------------- aggregation (self loops first)
__global__ void k_selfinit(const float* __restrict__ h, const float* __restrict__ dinv,
                           float* __restrict__ agg, int n) {
    size_t i = (size_t)blockIdx.x * blockDim.x + threadIdx.x;   // over n*32 float4s
    if (i >= (size_t)n * 32) return;
    int row = (int)(i >> 5);
    int c4  = (int)(i & 31) * 4;
    float d = dinv[row];
    float s = d * d;
    float4 v = *(const float4*)(h + (size_t)row * 128 + c4);
    v.x *= s; v.y *= s; v.z *= s; v.w *= s;
    *(float4*)(agg + (size_t)row * 128 + c4) = v;
}

// one wave per edge: 32 lanes x float4 = full 128-float row; f32 atomics hit L2
__global__ void __launch_bounds__(256)
k_edge(const float* __restrict__ h, const float* __restrict__ dinv,
       const int* __restrict__ ei, float* __restrict__ agg, int ne) {
    int wid  = (blockIdx.x * blockDim.x + threadIdx.x) >> 5;
    int lane = threadIdx.x & 31;
    if (wid >= ne) return;
    int src = ei[wid];
    int dst = ei[ne + wid];
    float norm = dinv[src] * dinv[dst];
    float4 v = *(const float4*)(h + (size_t)src * 128 + lane * 4);
    float* ad = agg + (size_t)dst * 128 + lane * 4;
    __hip_atomic_fetch_add(ad + 0, v.x * norm, __ATOMIC_RELAXED, __HIP_MEMORY_SCOPE_AGENT);
    __hip_atomic_fetch_add(ad + 1, v.y * norm, __ATOMIC_RELAXED, __HIP_MEMORY_SCOPE_AGENT);
    __hip_atomic_fetch_add(ad + 2, v.z * norm, __ATOMIC_RELAXED, __HIP_MEMORY_SCOPE_AGENT);
    __hip_atomic_fetch_add(ad + 3, v.w * norm, __ATOMIC_RELAXED, __HIP_MEMORY_SCOPE_AGENT);
}

__global__ void k_biasrelu(const float* __restrict__ agg, const float* __restrict__ b,
                           float* __restrict__ h, int n) {
    size_t i = (size_t)blockIdx.x * blockDim.x + threadIdx.x;
    if (i >= (size_t)n * 32) return;
    int row = (int)(i >> 5);
    int c4  = (int)(i & 31) * 4;
    float4 v = *(const float4*)(agg + (size_t)row * 128 + c4);
    v.x = fmaxf(v.x + b[c4 + 0], 0.f);
    v.y = fmaxf(v.y + b[c4 + 1], 0.f);
    v.z = fmaxf(v.z + b[c4 + 2], 0.f);
    v.w = fmaxf(v.w + b[c4 + 3], 0.f);
    *(float4*)(h + (size_t)row * 128 + c4) = v;
}

// ------------------------------------- categorical sample: 1 wave/row, 1 lane/action
__global__ void __launch_bounds__(256)
k_sample(const float* __restrict__ logits, float* __restrict__ act_out,
         float* __restrict__ logp_out, int n) {
    int wid  = (blockIdx.x * blockDim.x + threadIdx.x) >> 5;
    int lane = threadIdx.x & 31;
    if (wid >= n) return;
    float l = logits[(size_t)wid * 32 + lane];

    float m = l;
#pragma unroll
    for (int o = 16; o > 0; o >>= 1) m = fmaxf(m, __shfl_xor(m, o, 32));
    float e = __expf(l - m);
    float s = e;
#pragma unroll
    for (int o = 16; o > 0; o >>= 1) s += __shfl_xor(s, o, 32);
    float logZ = m + __logf(s);

    // deterministic per-(row,action) hash -> Gumbel
    unsigned long long hsh = (((unsigned long long)wid << 5) | (unsigned)lane);
    hsh = hsh * 0x9E3779B97F4A7C15ull + 0xD1B54A32D192ED03ull;
    hsh ^= hsh >> 33; hsh *= 0xFF51AFD7ED558CCDull;
    hsh ^= hsh >> 33; hsh *= 0xC4CEB9FE1A85EC53ull;
    hsh ^= hsh >> 33;
    float u = (float)((hsh >> 40) + 1ull) * (1.0f / 16777218.0f);
    float g = -__logf(-__logf(u));

    float score = l + g;
    int   idx   = lane;
#pragma unroll
    for (int o = 16; o > 0; o >>= 1) {
        float so = __shfl_xor(score, o, 32);
        int   io = __shfl_xor(idx,   o, 32);
        if (so > score || (so == score && io < idx)) { score = so; idx = io; }
    }
    float lsel = __shfl(l, idx, 32);
    if (lane == 0) {
        act_out[wid]  = (float)idx;
        logp_out[wid] = lsel - logZ;
    }
}

// single block -> deterministic sum of per-row log-probs
__global__ void k_reduce(const float* __restrict__ logp, float* __restrict__ out, int n) {
    __shared__ float sm[256];
    float s = 0.f;
    for (int i = threadIdx.x; i < n; i += 256) s += logp[i];
    sm[threadIdx.x] = s;
    __syncthreads();
    for (int st = 128; st > 0; st >>= 1) {
        if (threadIdx.x < st) sm[threadIdx.x] += sm[threadIdx.x + st];
        __syncthreads();
    }
    if (threadIdx.x == 0) out[0] = sm[0];
}

// --------------------------------------------------------------------- launch
extern "C" void kernel_launch(void* const* d_in, const int* in_sizes, int n_in,
                              void* d_out, int out_size, void* d_ws, size_t ws_size,
                              hipStream_t stream) {
    const float* x  = (const float*)d_in[0];
    const int*   ei = (const int*)  d_in[1];
    const float* W1 = (const float*)d_in[2];
    const float* b1 = (const float*)d_in[3];
    const float* W2 = (const float*)d_in[4];
    const float* b2 = (const float*)d_in[5];
    const float* Wh = (const float*)d_in[6];
    const float* bh = (const float*)d_in[7];

    const int n  = in_sizes[0] / 128;   // 100000
    const int ne = in_sizes[1] / 2;     // 1600000

    float* bufA = (float*)d_ws;                      // n*128
    float* bufB = bufA + (size_t)n * 128;            // n*128
    float* dinv = bufB + (size_t)n * 128;            // n
    float* logp = dinv + n;                          // n

    float* logits = (float*)d_out;                   // n*32
    float* act    = logits + (size_t)n * 32;         // n
    float* lp_out = act + n;                         // 1

    const int tpb = 256;
    const int gN   = (n + tpb - 1) / tpb;
    const int gE   = (ne + tpb - 1) / tpb;
    const int gRow = (n + 127) / 128;                // GEMM: 128 rows / block
    const int gEl  = (int)(((size_t)n * 32 + tpb - 1) / tpb);
    const int gW   = (n + 7) / 8;                    // 1 wave per row, 8 waves/blk
    const int gEW  = (ne + 7) / 8;

    // degrees -> dinv
    k_fill1   <<<gN, tpb, 0, stream>>>(dinv, n);
    k_degcount<<<gE, tpb, 0, stream>>>(ei, dinv, ne);
    k_rsqrt   <<<gN, tpb, 0, stream>>>(dinv, n);

    // layer 1: h0 = x@W1 ; agg = selfloop + edges ; h1 = relu(agg + b1)
    k_gemm<128, false, false><<<gRow, tpb, 0, stream>>>(x, W1, nullptr, bufA, n);
    k_selfinit<<<gEl, tpb, 0, stream>>>(bufA, dinv, bufB, n);
    k_edge    <<<gEW, tpb, 0, stream>>>(bufA, dinv, ei, bufB, ne);
    k_biasrelu<<<gEl, tpb, 0, stream>>>(bufB, b1, bufA, n);

    // layer 2
    k_gemm<128, false, false><<<gRow, tpb, 0, stream>>>(bufA, W2, nullptr, bufB, n);
    k_selfinit<<<gEl, tpb, 0, stream>>>(bufB, dinv, bufA, n);
    k_edge    <<<gEW, tpb, 0, stream>>>(bufB, dinv, ei, bufA, ne);
    k_biasrelu<<<gEl, tpb, 0, stream>>>(bufA, b2, bufB, n);

    // head + sampling
    k_gemm<32, true, false><<<gRow, tpb, 0, stream>>>(bufB, Wh, bh, logits, n);
    k_sample<<<gW, tpb, 0, stream>>>(logits, act, logp, n);
    k_reduce<<<1, tpb, 0, stream>>>(logp, lp_out, n);
}
